// QLSTMModel_61512521613691
// MI455X (gfx1250) — compile-verified, matrix operand order checked by reading, not run on hardware
//
#include <hip/hip_runtime.h>
#include <hip/hip_bf16.h>
#include <math.h>

typedef __attribute__((ext_vector_type(16))) _Float16 v16h;
typedef __attribute__((ext_vector_type(8)))  float    v8f;

union AF { v16h v; float4 q[2]; };

// ---------------- geometry ----------------
// B=1024, T=128, IN=8, H=128, NQ=8, DIM=256, NVL=4
constexpr int MT   = 16;    // batch rows per workgroup
constexpr int PRS  = 132;   // f32 LDS row stride (pad vs 64 banks)
constexpr int ENCS = 264;   // f16 LDS row stride (bytes=528, 16B aligned)

// shared-memory carve (bytes), all offsets 16B aligned
constexpr size_t OFF_H    = 0;
constexpr size_t OFF_C    = OFF_H    + MT*PRS*4;
constexpr size_t OFF_EMB  = OFF_C    + MT*PRS*4;
constexpr size_t OFF_PRE  = OFF_EMB  + MT*PRS*4;      // 4 gates x 16 x PRS f32
constexpr size_t OFF_ENC  = OFF_PRE  + 4*MT*PRS*4;
constexpr size_t OFF_SQ   = OFF_ENC  + MT*ENCS*2;
constexpr size_t OFF_IPW  = OFF_SQ   + MT*ENCS*2;     // 8x256 f32
constexpr size_t OFF_EMBW = OFF_IPW  + 8*256*4;       // 128x8 f32
constexpr size_t OFF_EMBB = OFF_EMBW + 128*8*4;
constexpr size_t OFF_EMBG = OFF_EMBB + 512;
constexpr size_t OFF_EMBBT= OFF_EMBG + 512;
constexpr size_t OFF_BG   = OFF_EMBBT+ 512;           // 4x128 gate biases
constexpr size_t OFF_ONG  = OFF_BG   + 2048;
constexpr size_t OFF_ONB  = OFF_ONG  + 512;
constexpr size_t OFF_OUTW = OFF_ONB  + 512;
constexpr size_t OFF_IPB  = OFF_OUTW + 512;           // 8 f32
constexpr size_t OFF_ING  = OFF_IPB  + 32;
constexpr size_t OFF_INB  = OFF_ING  + 32;
constexpr size_t OFF_RS   = OFF_INB  + 32;            // 16x16 f32 reduce
constexpr size_t OFF_RQ   = OFF_RS   + 1024;
constexpr size_t OFF_PROJ = OFF_RQ   + 1024;          // 16x8 f32
constexpr size_t OFF_AQ   = OFF_PROJ + 512;           // 16x8x2 f32
constexpr size_t SMEM_MAIN= OFF_AQ   + 1024;

// ======================================================================
// Precompute 1: U_g = fixed circuit (4 layers of CNOTs + RY(weights)) as a
// 256x256 matrix, packed into WMMA-B f16 fragment layout.
// B-frag layout assumption (mirror of ISA 16-bit A layout, M->N):
//   lane l: n = l&15, hi = l>>4; half j<8 -> k = kt*32+hi*8+j,
//   half j>=8 -> k = kt*32+16+hi*8+(j-8).
// ======================================================================
__global__ void qlstm_build_U(const float* __restrict__ w0, const float* __restrict__ w1,
                              const float* __restrict__ w2, const float* __restrict__ w3,
                              _Float16* __restrict__ Upk)
{
    extern __shared__ float M[];          // 256*256 f32 = 256 KB
    const int g = blockIdx.x;
    const int t = threadIdx.x;            // owns column t
    const float* w = (g==0) ? w0 : (g==1) ? w1 : (g==2) ? w2 : w3;

    for (int s = 0; s < 256; ++s) M[s*256 + t] = (s == t) ? 1.f : 0.f;

    for (int l = 0; l < 4; ++l) {
        // CNOT ladders: ctrl = 0,2,4,6 then 1,3,5 ; ctrl qubit q <-> bit 7-q
        for (int pass = 0; pass < 2; ++pass) {
            for (int ctrl = pass; ctrl < 7; ctrl += 2) {
                const int pc = 7 - ctrl, pt = 6 - ctrl;   // ctrl bit, target bit
                for (int m = 0; m < 64; ++m) {
                    const int low  = m & ((1 << pt) - 1);
                    const int high = (m >> pt) << (pc + 1);
                    const int s1 = high | (1 << pc) | low;  // ctrl=1, tgt=0
                    const int s2 = s1 | (1 << pt);          // ctrl=1, tgt=1
                    float tmp = M[s1*256 + t];
                    M[s1*256 + t] = M[s2*256 + t];
                    M[s2*256 + t] = tmp;
                }
            }
        }
        // RY(weights[l][q]) on every qubit
        for (int q = 0; q < 8; ++q) {
            const float th = 0.5f * w[l*8 + q];
            const float cc = cosf(th), ss = sinf(th);
            const int p = 7 - q;
            for (int m = 0; m < 128; ++m) {
                const int s0 = ((m >> p) << (p + 1)) | (m & ((1 << p) - 1));
                const int s1 = s0 | (1 << p);
                const float a0 = M[s0*256 + t], a1 = M[s1*256 + t];
                M[s0*256 + t] = cc*a0 - ss*a1;
                M[s1*256 + t] = ss*a0 + cc*a1;
            }
        }
    }
    __syncthreads();
    // pack B[k][n] = U[n][k] = M[n*256+k]  (final = enc @ U^T)
    for (int idx = t; idx < 65536; idx += 256) {
        const int j    = idx & 15;
        const int lane = (idx >> 4) & 31;
        const int nt   = (idx >> 9) & 15;
        const int kt   = idx >> 13;
        const int n    = nt*16 + (lane & 15);
        const int hi   = lane >> 4;
        const int k    = kt*32 + hi*8 + (j & 7) + ((j & 8) ? 16 : 0);
        Upk[(size_t)g*65536 + idx] = (_Float16)M[n*256 + k];
    }
}

// ======================================================================
// Precompute 2: W_g[h][s] = sum_q P_g[h][q] * sigma(q,s),
// sigma(q,s) = 1 - 2*bit_{7-q}(s). pre = final^2 @ W^T, so B2[k=s][n=h]=W[h][s].
// ======================================================================
__global__ void qlstm_build_W(const float* __restrict__ p0, const float* __restrict__ p1,
                              const float* __restrict__ p2, const float* __restrict__ p3,
                              _Float16* __restrict__ Wpk)
{
    const int g = blockIdx.x;
    const float* P = (g==0) ? p0 : (g==1) ? p1 : (g==2) ? p2 : p3;
    for (int idx = threadIdx.x; idx < 32768; idx += 256) {
        const int j    = idx & 15;
        const int lane = (idx >> 4) & 31;
        const int nt   = (idx >> 9) & 7;
        const int kt   = idx >> 12;
        const int n    = nt*16 + (lane & 15);
        const int hi   = lane >> 4;
        const int k    = kt*32 + hi*8 + (j & 7) + ((j & 8) ? 16 : 0);
        float acc = 0.f;
        #pragma unroll
        for (int q = 0; q < 8; ++q) {
            const float sgn = ((k >> (7 - q)) & 1) ? -1.f : 1.f;
            acc += P[n*8 + q] * sgn;
        }
        Wpk[(size_t)g*32768 + idx] = (_Float16)acc;
    }
}

// ======================================================================
// Main persistent kernel: 64 workgroups x 256 threads (8 wave32).
// Each workgroup owns 16 batch rows and runs the full T=128 recurrence.
// ======================================================================
__global__ __launch_bounds__(256)
void qlstm_scan(const float* __restrict__ x,    const float* __restrict__ pe,
                const float* __restrict__ embw, const float* __restrict__ embb,
                const float* __restrict__ embg, const float* __restrict__ embbt,
                const float* __restrict__ ipw,  const float* __restrict__ ipb,
                const float* __restrict__ ing,  const float* __restrict__ inb,
                const float* __restrict__ pib,  const float* __restrict__ pfb,
                const float* __restrict__ pgb,  const float* __restrict__ pob,
                const float* __restrict__ ong,  const float* __restrict__ onb,
                const float* __restrict__ outw, const float* __restrict__ outb,
                const _Float16* __restrict__ Upk, const _Float16* __restrict__ Wpk,
                float* __restrict__ out)
{
    extern __shared__ char smem[];
    float    * sh   = (float*)(smem + OFF_H);
    float    * sc   = (float*)(smem + OFF_C);
    float    * semb = (float*)(smem + OFF_EMB);
    float    * spre = (float*)(smem + OFF_PRE);
    _Float16 * senc = (_Float16*)(smem + OFF_ENC);
    _Float16 * ssq  = (_Float16*)(smem + OFF_SQ);
    float    * sipw = (float*)(smem + OFF_IPW);
    float    * sembw= (float*)(smem + OFF_EMBW);
    float    * sembb= (float*)(smem + OFF_EMBB);
    float    * sembg= (float*)(smem + OFF_EMBG);
    float    * sembbt=(float*)(smem + OFF_EMBBT);
    float    * sbg  = (float*)(smem + OFF_BG);
    float    * song = (float*)(smem + OFF_ONG);
    float    * sonb = (float*)(smem + OFF_ONB);
    float    * soutw= (float*)(smem + OFF_OUTW);
    float    * sipb = (float*)(smem + OFF_IPB);
    float    * sing = (float*)(smem + OFF_ING);
    float    * sinb = (float*)(smem + OFF_INB);
    float    * srs  = (float*)(smem + OFF_RS);
    float    * srq  = (float*)(smem + OFF_RQ);
    float    * sproj= (float*)(smem + OFF_PROJ);
    float    * saq  = (float*)(smem + OFF_AQ);

    const int tid = threadIdx.x;
    const int wv  = tid >> 5;
    const int ln  = tid & 31;
    const int m0  = blockIdx.x * MT;
    const int r4  = tid >> 4;      // row 0..15
    const int cg  = tid & 15;      // 16 col-groups of 8

    // cache parameters
    for (int i = tid; i < 2048; i += 256) sipw[i] = ipw[i];
    for (int i = tid; i < 1024; i += 256) sembw[i] = embw[i];
    if (tid < 128) {
        sembb[tid]  = embb[tid];  sembg[tid] = embg[tid]; sembbt[tid] = embbt[tid];
        sbg[tid]       = pib[tid]; sbg[128 + tid] = pfb[tid];
        sbg[256 + tid] = pgb[tid]; sbg[384 + tid] = pob[tid];
        song[tid] = ong[tid]; sonb[tid] = onb[tid]; soutw[tid] = outw[tid];
    }
    if (tid < 8) { sipb[tid] = ipb[tid]; sing[tid] = ing[tid]; sinb[tid] = inb[tid]; }
    for (int i = tid; i < MT*PRS; i += 256) { sh[i] = 0.f; sc[i] = 0.f; }
    __syncthreads();

    const float outb0 = outb[0];

    for (int t = 0; t < 128; ++t) {
        // ---------- (a) embedding tile: LN(x @ emb_w^T + b) + pe[t] ----------
        float ev[8];
        {
            const float* xr = x + ((size_t)(m0 + r4)*128 + t)*8;
            float xv[8];
            #pragma unroll
            for (int k = 0; k < 8; ++k) xv[k] = xr[k];
            float s = 0.f, q = 0.f;
            #pragma unroll
            for (int u = 0; u < 8; ++u) {
                const int c = cg*8 + u;
                float acc = sembb[c];
                #pragma unroll
                for (int k = 0; k < 8; ++k) acc += xv[k]*sembw[c*8 + k];
                ev[u] = acc; s += acc; q += acc*acc;
            }
            srs[r4*16 + cg] = s; srq[r4*16 + cg] = q;
        }
        __syncthreads();
        {
            float ts = 0.f, tq = 0.f;
            #pragma unroll
            for (int i = 0; i < 16; ++i) { ts += srs[r4*16 + i]; tq += srq[r4*16 + i]; }
            const float mean = ts*(1.f/128.f);
            const float inv  = rsqrtf(tq*(1.f/128.f) - mean*mean + 1e-5f);
            #pragma unroll
            for (int u = 0; u < 8; ++u) {
                const int c = cg*8 + u;
                semb[r4*PRS + c] = (ev[u] - mean)*inv*sembg[c] + sembbt[c] + pe[t*128 + c];
            }
        }
        __syncthreads();

        // ---------- (b) proj = LN(tanh([emb,h] @ ip_w^T + b)) -> qubit amps ----------
        if (tid < 128) {
            const int r = tid >> 3, q = tid & 7;
            float acc = sipb[q];
            for (int k = 0; k < 128; ++k) acc += semb[r*PRS + k]*sipw[q*256 + k];
            for (int k = 0; k < 128; ++k) acc += sh[r*PRS + k]*sipw[q*256 + 128 + k];
            sproj[r*8 + q] = tanhf(acc);
        }
        __syncthreads();
        if (tid < 128) {
            const int r = tid >> 3, q = tid & 7;
            float m = 0.f, vv = 0.f;
            #pragma unroll
            for (int j = 0; j < 8; ++j) { const float p = sproj[r*8 + j]; m += p; vv += p*p; }
            m *= 0.125f; vv = vv*0.125f - m*m;
            const float val = (sproj[r*8 + q] - m)*rsqrtf(vv + 1e-5f)*sing[q] + sinb[q];
            const float cc = cosf(0.5f*val), ss = sinf(0.5f*val);
            saq[(r*8 + q)*2 + 0] = (cc - ss)*0.70710678118654752f;
            saq[(r*8 + q)*2 + 1] = (cc + ss)*0.70710678118654752f;
        }
        __syncthreads();

        // ---------- (c) product-state encoding: enc[r][s] = prod_q a(bit) ----------
        {
            const int r = tid >> 4, sl = tid & 15;
            #pragma unroll
            for (int u = 0; u < 16; ++u) {
                const int s = sl*16 + u;
                float amp = 1.f;
                #pragma unroll
                for (int q = 0; q < 8; ++q)
                    amp *= saq[(r*8 + q)*2 + ((s >> (7 - q)) & 1)];
                senc[r*ENCS + s] = (_Float16)amp;
            }
        }
        __syncthreads();

        // ---------- (d) per gate: WMMA GEMM1 (enc@U^T), square, GEMM2 (sq@W^T) ----------
        const int ar = ln & 15, ahi = ln >> 4;
        #pragma unroll 1
        for (int g = 0; g < 4; ++g) {
            // GEMM1: wave handles N-tiles wv and wv+8 of the 256-col output
            v8f acc0 = {}; v8f acc1 = {};
            for (int kt = 0; kt < 8; ++kt) {
                AF a;
                a.q[0] = *(const float4*)(senc + ar*ENCS + kt*32 + ahi*8);
                a.q[1] = *(const float4*)(senc + ar*ENCS + kt*32 + 16 + ahi*8);
                const _Float16* bb0 = Upk + (((size_t)g*8 + kt)*16 + wv)*512 + ln*16;
                const _Float16* bb1 = bb0 + (size_t)8*512;
                AF b0; b0.q[0] = *(const float4*)(bb0); b0.q[1] = *(const float4*)(bb0 + 8);
                AF b1; b1.q[0] = *(const float4*)(bb1); b1.q[1] = *(const float4*)(bb1 + 8);
                acc0 = __builtin_amdgcn_wmma_f32_16x16x32_f16(false, a.v, false, b0.v, (short)0, acc0, false, false);
                acc1 = __builtin_amdgcn_wmma_f32_16x16x32_f16(false, a.v, false, b1.v, (short)0, acc1, false, false);
            }
            {   // square amplitudes -> sq tile (f16)
                const int row = (ln >> 4)*8, col = ln & 15;
                #pragma unroll
                for (int i = 0; i < 8; ++i) {
                    const float v0 = acc0[i], v1 = acc1[i];
                    ssq[(row + i)*ENCS + wv*16 + col]       = (_Float16)(v0*v0);
                    ssq[(row + i)*ENCS + (wv + 8)*16 + col] = (_Float16)(v1*v1);
                }
            }
            __syncthreads();
            // GEMM2: wave handles N-tile wv of the 128-col output
            v8f acc = {};
            for (int kt = 0; kt < 8; ++kt) {
                AF a;
                a.q[0] = *(const float4*)(ssq + ar*ENCS + kt*32 + ahi*8);
                a.q[1] = *(const float4*)(ssq + ar*ENCS + kt*32 + 16 + ahi*8);
                const _Float16* bb = Wpk + (((size_t)g*8 + kt)*8 + wv)*512 + ln*16;
                AF b; b.q[0] = *(const float4*)(bb); b.q[1] = *(const float4*)(bb + 8);
                acc = __builtin_amdgcn_wmma_f32_16x16x32_f16(false, a.v, false, b.v, (short)0, acc, false, false);
            }
            {
                const int row = (ln >> 4)*8, col = wv*16 + (ln & 15);
                #pragma unroll
                for (int i = 0; i < 8; ++i)
                    spre[g*MT*PRS + (row + i)*PRS + col] = acc[i];
            }
            __syncthreads();
        }

        // ---------- (e) LSTM pointwise + output LN + 128->1 projection ----------
        {
            float hv[8]; float s = 0.f, q = 0.f;
            #pragma unroll
            for (int u = 0; u < 8; ++u) {
                const int c = cg*8 + u;
                float iv = spre[0*MT*PRS + r4*PRS + c] + sbg[c];
                float fv = spre[1*MT*PRS + r4*PRS + c] + sbg[128 + c];
                float gv = spre[2*MT*PRS + r4*PRS + c] + sbg[256 + c];
                float ov = spre[3*MT*PRS + r4*PRS + c] + sbg[384 + c];
                iv = 1.f/(1.f + expf(-iv));
                fv = 1.f/(1.f + expf(-fv));
                gv = tanhf(gv);
                ov = 1.f/(1.f + expf(-ov));
                const float cn = fv*sc[r4*PRS + c] + iv*gv;
                const float hn = ov*tanhf(cn);
                sc[r4*PRS + c] = cn; sh[r4*PRS + c] = hn;
                const float sr = hn + semb[r4*PRS + c];
                hv[u] = sr; s += sr; q += sr*sr;
            }
            srs[r4*16 + cg] = s; srq[r4*16 + cg] = q;
            __syncthreads();
            float ts = 0.f, tq = 0.f;
            #pragma unroll
            for (int i = 0; i < 16; ++i) { ts += srs[r4*16 + i]; tq += srq[r4*16 + i]; }
            const float mean = ts*(1.f/128.f);
            const float inv  = rsqrtf(tq*(1.f/128.f) - mean*mean + 1e-5f);
            __syncthreads();   // protect srs before reuse for dot reduction
            float dacc = 0.f;
            #pragma unroll
            for (int u = 0; u < 8; ++u) {
                const int c = cg*8 + u;
                const float ov = (hv[u] - mean)*inv*song[c] + sonb[c];
                dacc += ov*soutw[c];
            }
            srs[r4*16 + cg] = dacc;
            __syncthreads();
            if (cg == 0) {
                float tot = outb0;
                #pragma unroll
                for (int i = 0; i < 16; ++i) tot += srs[r4*16 + i];
                out[(size_t)(m0 + r4)*128 + t] = tot;
            }
        }
        __syncthreads();
    }
}

// ======================================================================
extern "C" void kernel_launch(void* const* d_in, const int* in_sizes, int n_in,
                              void* d_out, int out_size, void* d_ws, size_t ws_size,
                              hipStream_t stream)
{
    (void)in_sizes; (void)n_in; (void)out_size; (void)ws_size;
    const float* x     = (const float*)d_in[0];
    const float* pe    = (const float*)d_in[1];
    const float* emb_w = (const float*)d_in[2];
    const float* emb_b = (const float*)d_in[3];
    const float* emb_g = (const float*)d_in[4];
    const float* emb_bt= (const float*)d_in[5];
    const float* ip_w  = (const float*)d_in[6];
    const float* ip_b  = (const float*)d_in[7];
    const float* in_g  = (const float*)d_in[8];
    const float* in_b  = (const float*)d_in[9];
    const float* wq_i  = (const float*)d_in[10];
    const float* wq_f  = (const float*)d_in[11];
    const float* wq_g  = (const float*)d_in[12];
    const float* wq_o  = (const float*)d_in[13];
    const float* pi_w  = (const float*)d_in[14];
    const float* pi_b  = (const float*)d_in[15];
    const float* pf_w  = (const float*)d_in[16];
    const float* pf_b  = (const float*)d_in[17];
    const float* pg_w  = (const float*)d_in[18];
    const float* pg_b  = (const float*)d_in[19];
    const float* po_w  = (const float*)d_in[20];
    const float* po_b  = (const float*)d_in[21];
    const float* on_g  = (const float*)d_in[22];
    const float* on_b  = (const float*)d_in[23];
    const float* out_w = (const float*)d_in[24];
    const float* out_b = (const float*)d_in[25];

    _Float16* Upk = (_Float16*)d_ws;            // 4 * 256*256 f16 = 512 KB
    _Float16* Wpk = Upk + (size_t)4*65536;      // 4 * 256*128 f16 = 256 KB

    qlstm_build_U<<<dim3(4), dim3(256), 256*256*sizeof(float), stream>>>(
        wq_i, wq_f, wq_g, wq_o, Upk);
    qlstm_build_W<<<dim3(4), dim3(256), 0, stream>>>(
        pi_w, pf_w, pg_w, po_w, Wpk);
    qlstm_scan<<<dim3(64), dim3(256), SMEM_MAIN, stream>>>(
        x, pe, emb_w, emb_b, emb_g, emb_bt, ip_w, ip_b, in_g, in_b,
        pi_b, pf_b, pg_b, po_b, on_g, on_b, out_w, out_b,
        Upk, Wpk, (float*)d_out);
}
